// NTXentLossGlobal_26104811225348
// MI455X (gfx1250) — compile-verified, matrix operand order checked by reading, not run on hardware
//
#include <hip/hip_runtime.h>

typedef __attribute__((ext_vector_type(16))) _Float16 v16h;
typedef __attribute__((ext_vector_type(8)))  _Float16 v8h;
typedef __attribute__((ext_vector_type(8)))  float    v8f;

namespace {
constexpr int   kB            = 4096;
constexpr int   kD            = 128;
constexpr int   kN            = 2 * kB;            // 8192 rows/cols of sim
constexpr int   kTiles        = kN / 16;           // 512 16-wide tiles
constexpr int   kTilesPerGrp  = kTiles / 2;        // 256 column tiles per group
constexpr int   kRowTilesBlk  = 4;                 // row tiles per block
constexpr int   kBlockThreads = 256;               // 8 waves: 4 row tiles x 2 col groups
constexpr int   kBlocks       = kTiles / kRowTilesBlk;  // 128
constexpr int   kLdsStride    = 136;               // halves per LDS row (128 + 8 pad)
constexpr float kAlpha        = 7.21347520444482f; // log2(e) / TEMPERATURE(0.2)
constexpr float kLn2          = 0.6931471805599453f;
}

__global__ void ntxent_zero(float* out) { out[0] = 0.0f; }

// One-time f32 -> f16 conversion of z = concat(z1, z2) into workspace (unscaled:
// the 1/T * log2(e) factor is folded into the A-side fragments only).
__launch_bounds__(256)
__global__ void ntxent_cvt(const float* __restrict__ z1,
                           const float* __restrict__ z2,
                           _Float16* __restrict__ zh)
{
  const int chunk = blockIdx.x * blockDim.x + threadIdx.x;  // 8-elem chunks
  const int e0    = chunk * 8;
  const float* src = (e0 < kB * kD) ? (z1 + e0) : (z2 + (e0 - kB * kD));
  v8f f = *(const v8f*)src;
  v8h hv;
#pragma unroll
  for (int e = 0; e < 8; ++e) hv[e] = (_Float16)f[e];
  *(v8h*)(zh + e0) = hv;
}

__launch_bounds__(kBlockThreads)
__global__ void ntxent_fused(const float* __restrict__ z1,
                             const float* __restrict__ z2,
                             const _Float16* __restrict__ zh,
                             float* __restrict__ out)
{
  // [col-group][ping-pong] staging tiles + reduction scratch
  __shared__ _Float16 btile[2][2][16 * kLdsStride];
  __shared__ float    stats[2][64][3];   // [group][row-in-block][m,s,tgt]
  __shared__ float    lsum[64];

  const int lane = threadIdx.x & 31;
  const int wave = threadIdx.x >> 5;
  const int g    = wave >> 2;                       // column group (0 or 1)
  const int r    = blockIdx.x * kRowTilesBlk + (wave & 3);  // this wave's row tile
  const int h    = lane >> 4;
  const int n16  = lane & 15;

  // ---- async-staging addressing (per thread: one 16B chunk of each tile) ----
  const int srow = threadIdx.x >> 4;                // 0..15 row within tile
  const int sc8  = threadIdx.x & 15;                // 16B chunk within row
  const int goff = (srow * kD + sc8 * 8) * 2;       // byte offset inside a tile
  const unsigned ldsa[2][2] = {
    { (unsigned)(uintptr_t)&btile[0][0][srow * kLdsStride + sc8 * 8],
      (unsigned)(uintptr_t)&btile[0][1][srow * kLdsStride + sc8 * 8] },
    { (unsigned)(uintptr_t)&btile[1][0][srow * kLdsStride + sc8 * 8],
      (unsigned)(uintptr_t)&btile[1][1][srow * kLdsStride + sc8 * 8] } };

  // ---- A fragments: my 16 rows x 128 cols, f32 -> f16, scaled by log2(e)/T ----
  // 16-bit A 16x32 layout (ISA 7.12.2): lane holds row (lane&15); K runs
  // [8h..8h+7] and [16+8h..16+8h+7] within each 32-wide K chunk.
  v16h afrag[4];
  {
    const int ra = r * 16 + n16;
    const float* zr = (ra < kB) ? (z1 + (size_t)ra * kD)
                                : (z2 + (size_t)(ra - kB) * kD);
#pragma unroll
    for (int kc = 0; kc < 4; ++kc) {
      v8f f0 = *(const v8f*)(zr + kc * 32 + h * 8);
      v8f f1 = *(const v8f*)(zr + kc * 32 + 16 + h * 8);
      v16h a;
#pragma unroll
      for (int e = 0; e < 8; ++e) {
        a[e]     = (_Float16)(kAlpha * f0[e]);
        a[e + 8] = (_Float16)(kAlpha * f1[e]);
      }
      afrag[kc] = a;
    }
  }

  // ---- preload tile i=0 for both groups into ping buffer 0 ----
  {
    const uint64_t sb0 = (uint64_t)(uintptr_t)zh;
    const uint64_t sb1 = (uint64_t)(uintptr_t)zh +
                         (uint64_t)kTilesPerGrp * 16 * kD * 2;
    asm volatile("global_load_async_to_lds_b128 %0, %1, %2 offset:0"
                 :: "v"(ldsa[0][0]), "v"(goff), "s"(sb0) : "memory");
    asm volatile("global_load_async_to_lds_b128 %0, %1, %2 offset:0"
                 :: "v"(ldsa[1][0]), "v"(goff), "s"(sb1) : "memory");
    asm volatile("s_wait_asynccnt 0x0" ::: "memory");
  }
  __syncthreads();

  // Online-softmax state per C-matrix slot j:
  // lanes 0-15 track row 16r+j, lanes 16-31 track row 16r+j+8.
  float m[8], s[8], tg[8];
#pragma unroll
  for (int j = 0; j < 8; ++j) { m[j] = -1.0e30f; s[j] = 0.0f; tg[j] = 0.0f; }

  const int pt = (r < kTiles / 2) ? r + kTiles / 2 : r - kTiles / 2;

#pragma unroll 2   // makes cur/nxt compile-time: no cndmask buffer selects
  for (int i = 0; i < kTilesPerGrp; ++i) {
    const int cur = i & 1;
    const int nxt = cur ^ 1;
    const int c   = g * kTilesPerGrp + i;           // this wave's column tile

    // ---- issue async L2->LDS fill for the next iteration (ping-pong) ----
    {
      const int inext = (i + 1) & (kTilesPerGrp - 1);
      const uint64_t sb0 = (uint64_t)(uintptr_t)zh +
                           (uint64_t)inext * 16 * kD * 2;
      const uint64_t sb1 = (uint64_t)(uintptr_t)zh +
                           (uint64_t)(kTilesPerGrp + inext) * 16 * kD * 2;
      asm volatile("global_load_async_to_lds_b128 %0, %1, %2 offset:0"
                   :: "v"(ldsa[0][nxt]), "v"(goff), "s"(sb0) : "memory");
      asm volatile("global_load_async_to_lds_b128 %0, %1, %2 offset:0"
                   :: "v"(ldsa[1][nxt]), "v"(goff), "s"(sb1) : "memory");
    }

    // ---- load ALL B fragments first (batch the 8 ds_load_b128) ----
    v16h bf[4];
    {
      const _Float16* base = &btile[g][cur][n16 * kLdsStride];
#pragma unroll
      for (int kc = 0; kc < 4; ++kc) {
        v8h b0 = *(const v8h*)(base + kc * 32 + h * 8);
        v8h b1 = *(const v8h*)(base + kc * 32 + 16 + h * 8);
        v16h b;
#pragma unroll
        for (int e = 0; e < 8; ++e) { b[e] = b0[e]; b[e + 8] = b1[e]; }
        bf[kc] = b;
      }
    }

    // ---- 16x16x128 via 4 chained WMMAs ----
    v8f acc = {};
#pragma unroll
    for (int kc = 0; kc < 4; ++kc)
      acc = __builtin_amdgcn_wmma_f32_16x16x32_f16(
          false, afrag[kc], false, bf[kc], (short)0, acc, false, false);

    // ---- diagonal mask (self-similarity excluded from softmax) ----
    if (c == r) {
#pragma unroll
      for (int j = 0; j < 8; ++j)
        if (n16 == j + 8 * h) acc[j] = -3.0e38f;
    }
    // ---- capture positive-pair logit ----
    if (c == pt) {
#pragma unroll
      for (int j = 0; j < 8; ++j)
        if (n16 == j + 8 * h) tg[j] = acc[j];
    }

    // ---- online softmax: one exp2 per element, |.|/neg as src modifiers ----
#pragma unroll
    for (int j = 0; j < 8; ++j) {
      const float x = acc[j];
      const float d = __builtin_amdgcn_exp2f(-__builtin_fabsf(x - m[j]));
      m[j] = fmaxf(m[j], x);
      s[j] = (x > m[j]) ? (s[j] * d + 1.0f) : (s[j] + d);
    }

    asm volatile("s_wait_asynccnt 0x0" ::: "memory");
    __syncthreads();
  }

  // ---- merge (m,s,tg) across the 16 lanes of each half-wave; dump to LDS ----
#pragma unroll
  for (int j = 0; j < 8; ++j) {
    float mm = m[j], ss = s[j], tt = tg[j];
#pragma unroll
    for (int off = 1; off < 16; off <<= 1) {
      const float mo = __shfl_xor(mm, off, 32);
      const float so = __shfl_xor(ss, off, 32);
      const float nm = fmaxf(mm, mo);
      ss = ss * __builtin_amdgcn_exp2f(mm - nm) +
           so * __builtin_amdgcn_exp2f(mo - nm);
      mm = nm;
      tt += __shfl_xor(tt, off, 32);
    }
    if (n16 == j) {  // lane j: rows 16r+j (h=0) / 16r+j+8 (h=1)
      const int row = (wave & 3) * 16 + j + 8 * h;
      stats[g][row][0] = mm;
      stats[g][row][1] = ss;
      stats[g][row][2] = tt;
    }
  }
  __syncthreads();

  // ---- combine the two column-group partials per row; block reduction ----
  if (threadIdx.x < 64) {
    const float m0 = stats[0][threadIdx.x][0], s0 = stats[0][threadIdx.x][1];
    const float m1 = stats[1][threadIdx.x][0], s1 = stats[1][threadIdx.x][1];
    const float tt = stats[0][threadIdx.x][2] + stats[1][threadIdx.x][2];
    const float nm = fmaxf(m0, m1);
    const float ss = s0 * __builtin_amdgcn_exp2f(m0 - nm) +
                     s1 * __builtin_amdgcn_exp2f(m1 - nm);
    lsum[threadIdx.x] = nm + __builtin_amdgcn_logf(ss) - tt;
  }
  __syncthreads();
  if (threadIdx.x == 0) {
    float a = 0.0f;
    for (int i = 0; i < 64; ++i) a += lsum[i];
    atomicAdd(out, a * kLn2 / (float)kN);
  }
}

extern "C" void kernel_launch(void* const* d_in, const int* in_sizes, int n_in,
                              void* d_out, int out_size, void* d_ws, size_t ws_size,
                              hipStream_t stream) {
  (void)in_sizes; (void)n_in; (void)out_size; (void)ws_size;
  const float* z1 = (const float*)d_in[0];
  const float* z2 = (const float*)d_in[1];
  float*       out = (float*)d_out;
  _Float16*    zh  = (_Float16*)d_ws;   // 8192*128 halves = 2 MB staging

  ntxent_zero<<<1, 1, 0, stream>>>(out);
  ntxent_cvt<<<(kN * kD / 8) / 256, 256, 0, stream>>>(z1, z2, zh);
  ntxent_fused<<<kBlocks, kBlockThreads, 0, stream>>>(z1, z2, zh, out);
}
